// rlstm_44160853737667
// MI455X (gfx1250) — matrix operand order repaired and checked by reading, not compiled, not run on hardware
//
#include <hip/hip_runtime.h>
#include <hip/hip_bf16.h>
#include <stdint.h>

typedef __bf16 bf16_t;
typedef __attribute__((ext_vector_type(16))) __bf16 v16bf;
typedef __attribute__((ext_vector_type(8)))  __bf16 v8bf;
typedef __attribute__((ext_vector_type(8)))  float  v8f;

#define LSTM_H   512
#define NPROP    1024
#define SEQ      256
#define CIN      128

#if defined(__gfx1250__) && __has_builtin(__builtin_amdgcn_tensor_load_to_lds) && \
    __has_builtin(__builtin_amdgcn_s_wait_tensorcnt)
#define HAVE_TDM 1
typedef __attribute__((ext_vector_type(4))) unsigned int v4u;
typedef __attribute__((ext_vector_type(8))) int          v8i;
typedef __attribute__((ext_vector_type(4))) int          v4i;

// 2D tile load Global(bf16, row-major, arbitrary row stride) -> LDS (packed).
// D# per CDNA5 ISA ch.8: group0 {count, lds_addr, global_addr, type=2},
// group1 {data_size=2B, tensor dims (huge => no OOB clip), tile dims, stride}.
// clang-23 toolchain uses the 6-arg builtin: extra int32x8 operand (zeroed).
__device__ __forceinline__ void tdm_load_2d(
    uint32_t lds_byte, const bf16_t* gsrc,
    uint32_t tile_x, uint32_t tile_y, uint64_t row_stride_elems)
{
    const uint64_t ga = (uint64_t)(uintptr_t)gsrc;
    v4u g0;
    g0[0] = 1u;                                        // count=1, user descriptor
    g0[1] = lds_byte;                                  // lds_addr
    g0[2] = (uint32_t)ga;                              // global_addr[31:0]
    g0[3] = (uint32_t)((ga >> 32) & 0x01FFFFFFu) | (2u << 30); // addr[56:32] | type=2
    const uint32_t D0 = 1u << 30, D1 = 1u << 30;       // tensor dims: huge
    v8i g1;
    g1[0] = (int)(1u << 16);                           // data_size=1 => 2 bytes
    g1[1] = (int)((D0 & 0xFFFFu) << 16);               // tensor_dim0[15:0]
    g1[2] = (int)(((D0 >> 16) & 0xFFFFu) | ((D1 & 0xFFFFu) << 16));
    g1[3] = (int)(((D1 >> 16) & 0xFFFFu) | (tile_x << 16));    // tile_dim0
    g1[4] = (int)(tile_y & 0xFFFFu);                   // tile_dim1, tile_dim2=0
    g1[5] = (int)(uint32_t)(row_stride_elems & 0xFFFFFFFFu);   // dim0_stride lo32
    g1[6] = (int)(uint32_t)((row_stride_elems >> 32) & 0xFFFFu); // hi16
    g1[7] = 0;
    v4i g2;
    g2[0] = (int)(1u << 30);                           // tensor_dim2 (unused, big)
    g2[1] = 0; g2[2] = 0; g2[3] = 0;
    v4i g3 = (v4i)0;
    v8i g4 = (v8i)0;                                   // aux words (6-arg form)
    __builtin_amdgcn_tensor_load_to_lds(g0, g1, g2, g3, g4, 0);
}
#endif

__device__ __forceinline__ float sigm_f(float x) { return 1.0f / (1.0f + __expf(-x)); }
__device__ __forceinline__ float tanh_f(float x) { return 1.0f - 2.0f / (1.0f + __expf(2.0f * x)); }

__device__ __forceinline__ v8bf cvt8(const float* s) {
    const float4 a = *(const float4*)(s + 0);
    const float4 b = *(const float4*)(s + 4);
    v8bf p;
    p[0]=(bf16_t)a.x; p[1]=(bf16_t)a.y; p[2]=(bf16_t)a.z; p[3]=(bf16_t)a.w;
    p[4]=(bf16_t)b.x; p[5]=(bf16_t)b.y; p[6]=(bf16_t)b.z; p[7]=(bf16_t)b.w;
    return p;
}

// ---------------------------------------------------------------------------
// One LSTM timestep for one layer.  out(n,g) = A(n,:) . W(g,:) + bias(g),
// A = [x_t | h_prev], K = Kx + 512.  Block: 256 thr (8 waves), tile
// M=64 props x J=32 units x 4 gates; gates fused in-register -> c,h update.
// Double-buffered LDS; staging via per-wave TDM tensor_load_to_lds slices.
// ---------------------------------------------------------------------------
__global__ __launch_bounds__(256) void lstm_step_kernel(
    const float*  __restrict__ xF,        // layer0 x source (f32) or null
    const bf16_t* __restrict__ xB,        // layer1 x source (bf16) or null
    int xStride, int xOff,
    const bf16_t* __restrict__ hPrev,     // (1024 x 512) bf16
    const bf16_t* __restrict__ WihB,      // (2048 x Kx) bf16 row-major
    const bf16_t* __restrict__ WhhB,      // (2048 x 512) bf16 row-major
    const float*  __restrict__ bias,      // (2048) f32
    float*        __restrict__ cState,    // (1024 x 512) f32 in/out
    bf16_t*       __restrict__ hOutB,     // (1024 x 512) bf16 out
    float*        __restrict__ hOutF,     // (1024 x 512) f32 out (nullable)
    int Kx)
{
    __shared__ bf16_t Als[2][64 * 64];     // 2 x 8 KB
    __shared__ bf16_t Bls[2][128 * 64];    // 2 x 16 KB

    const int tid    = threadIdx.x;
    const int lane   = tid & 31;
    const int wave   = tid >> 5;
    const int mt     = wave & 3;
    const int jt     = wave >> 2;
    const int blockM = blockIdx.x;         // props / 64
    const int blockJ = blockIdx.y;         // units / 32

    v8f acc[4] = {};
    const int nCh = (Kx + LSTM_H) / 64;

    // ---- stage one 64-K chunk into LDS buffer b ----
    auto stage = [&](int cc, int b) {
        const int  k0  = cc * 64;
        const bool isX = (k0 < Kx);
#ifdef HAVE_TDM
        // B slice per wave: gate = wave>>1, col half = (wave&1)*16 -> 16 rows x 64 k
        {
            const int gate = wave >> 1;
            const int colh = (wave & 1) * 16;
            const int grow = gate * LSTM_H + blockJ * 32 + colh;
            const bf16_t* src = isX ? (WihB + (size_t)grow * Kx + k0)
                                    : (WhhB + (size_t)grow * LSTM_H + (k0 - Kx));
            const uint32_t lds = (uint32_t)(uintptr_t)&Bls[b][(gate * 32 + colh) * 64];
            tdm_load_2d(lds, src, 64u, 16u, (uint64_t)(isX ? Kx : LSTM_H));
        }
        // A slice per wave: rows [wave*8, wave*8+8) x 64 k
        if (isX && xF) {
            // f32 input -> convert while staging (plain path, layer0 first chunks)
            const int r   = wave * 8 + (lane >> 2);
            const int seg = (lane & 3) * 16;
            const float* s = xF + (size_t)(blockM * 64 + r) * xStride + xOff + k0 + seg;
            bf16_t* dst = &Als[b][r * 64 + seg];
            *(v8bf*)dst       = cvt8(s);
            *(v8bf*)(dst + 8) = cvt8(s + 8);
        } else {
            const int r0 = wave * 8;
            const bf16_t* src;
            uint64_t st;
            if (isX) { src = xB    + (size_t)(blockM * 64 + r0) * xStride + xOff + k0; st = (uint64_t)xStride; }
            else     { src = hPrev + (size_t)(blockM * 64 + r0) * LSTM_H + (k0 - Kx); st = LSTM_H; }
            tdm_load_2d((uint32_t)(uintptr_t)&Als[b][r0 * 64], src, 64u, 8u, st);
        }
#else
        // plain staging fallback
        {
            const int r = tid >> 2, seg = (tid & 3) * 16;
            const int prop = blockM * 64 + r;
            bf16_t* dst = &Als[b][r * 64 + seg];
            if (isX) {
                if (xF) {
                    const float* s = xF + (size_t)prop * xStride + xOff + k0 + seg;
                    *(v8bf*)dst = cvt8(s); *(v8bf*)(dst + 8) = cvt8(s + 8);
                } else {
                    const v8bf* s = (const v8bf*)(xB + (size_t)prop * xStride + xOff + k0 + seg);
                    *(v8bf*)dst = s[0]; *(v8bf*)(dst + 8) = s[1];
                }
            } else {
                const v8bf* s = (const v8bf*)(hPrev + (size_t)prop * LSTM_H + (k0 - Kx) + seg);
                *(v8bf*)dst = s[0]; *(v8bf*)(dst + 8) = s[1];
            }
        }
        {
            const int row = tid >> 1, kk = (tid & 1) * 32;
            const int grow = (row >> 5) * LSTM_H + blockJ * 32 + (row & 31);
            const bf16_t* s = isX ? (WihB + (size_t)grow * Kx + k0 + kk)
                                  : (WhhB + (size_t)grow * LSTM_H + (k0 - Kx + kk));
            bf16_t* dst = &Bls[b][row * 64 + kk];
            const v8bf* sv = (const v8bf*)s;
            *(v8bf*)(dst + 0)  = sv[0];
            *(v8bf*)(dst + 8)  = sv[1];
            *(v8bf*)(dst + 16) = sv[2];
            *(v8bf*)(dst + 24) = sv[3];
        }
#endif
    };

    // ---- consume one staged chunk: preload all fragments, then 8 WMMAs ----
    auto compute = [&](int b) {
        const int mrow  = mt * 16 + (lane & 15);
        const int khalf = (lane >> 4) * 16;
        const int ncol  = lane & 15;
        const int ksec  = (lane >> 4) * 16;
        v16bf af[2];
        v16bf bfr[2][4];
        #pragma unroll
        for (int ks = 0; ks < 2; ++ks) {
            const bf16_t* ap = &Als[b][mrow * 64 + ks * 32 + khalf];
            const v8bf a0 = *(const v8bf*)ap;
            const v8bf a1 = *(const v8bf*)(ap + 8);
            af[ks] = __builtin_shufflevector(a0, a1,
                0,1,2,3,4,5,6,7,8,9,10,11,12,13,14,15);
            #pragma unroll
            for (int g = 0; g < 4; ++g) {
                const bf16_t* bp = &Bls[b][(g * 32 + jt * 16 + ncol) * 64 + ks * 32 + ksec];
                const v8bf b0 = *(const v8bf*)bp;
                const v8bf b1 = *(const v8bf*)(bp + 8);
                bfr[ks][g] = __builtin_shufflevector(b0, b1,
                    0,1,2,3,4,5,6,7,8,9,10,11,12,13,14,15);
            }
        }
        #pragma unroll
        for (int ks = 0; ks < 2; ++ks)
            #pragma unroll
            for (int g = 0; g < 4; ++g)
                acc[g] = __builtin_amdgcn_wmma_f32_16x16x32_bf16(
                    false, af[ks], false, bfr[ks][g], (short)0, acc[g], false, false);
    };

    // ---- software pipeline over K chunks (double-buffered) ----
    stage(0, 0);
    for (int cc = 0; cc < nCh; ++cc) {
#ifdef HAVE_TDM
        __builtin_amdgcn_s_wait_tensorcnt(0);   // this wave's TDM slices landed
#endif
        __syncthreads();                         // chunk cc visible to all waves
        if (cc + 1 < nCh) stage(cc + 1, (cc + 1) & 1);
        compute(cc & 1);
        __syncthreads();                         // done reading before next overwrite
    }

    // ---- gate math + state update (C/D: M = r + 8*(L/16), N = L%16) ----
    const int jglob    = blockJ * 32 + jt * 16 + (lane & 15);
    const int propBase = blockM * 64 + mt * 16 + ((lane >> 4) * 8);
    const float bi = bias[jglob];
    const float bf = bias[LSTM_H + jglob];
    const float bg = bias[2 * LSTM_H + jglob];
    const float bo = bias[3 * LSTM_H + jglob];
    #pragma unroll
    for (int r = 0; r < 8; ++r) {
        const size_t idx = (size_t)(propBase + r) * LSTM_H + jglob;
        const float gi = acc[0][r] + bi;
        const float gf = acc[1][r] + bf;
        const float gg = acc[2][r] + bg;
        const float go = acc[3][r] + bo;
        const float cp = cState[idx];
        const float cn = sigm_f(gf) * cp + sigm_f(gi) * tanh_f(gg);
        const float hn = sigm_f(go) * tanh_f(cn);
        cState[idx] = cn;
        hOutB[idx]  = (bf16_t)hn;
        if (hOutF) hOutF[idx] = hn;
    }
}

// ---------------------------------------------------------------------------
__global__ __launch_bounds__(256) void cvt_f32_to_bf16(
    const float* __restrict__ in, bf16_t* __restrict__ out, int n)
{
    const int i = ((int)blockIdx.x * 256 + (int)threadIdx.x) * 8;
    if (i + 8 <= n) *(v8bf*)(out + i) = cvt8(in + i);
}

__global__ __launch_bounds__(64) void head_stage1(
    const float* __restrict__ h,      const float* __restrict__ Wcls,
    const float* __restrict__ bcls,   const float* __restrict__ Wbb,
    const float* __restrict__ bbb,    const int*   __restrict__ labels,
    const float* __restrict__ props,  float*       __restrict__ out3)
{
    __shared__ float logits[40];
    __shared__ float bb[2];
    const int p = blockIdx.x;
    const int j = threadIdx.x;
    const float* hp = h + (size_t)p * LSTM_H;
    if (j < 40) {
        const float* w = Wcls + (size_t)j * LSTM_H;
        float s = bcls[j];
        for (int k = 0; k < LSTM_H; ++k) s += hp[k] * w[k];
        logits[j] = s;
    } else if (j < 42) {
        const float* w = Wbb + (size_t)(j - 40) * LSTM_H;
        float s = bbb[j - 40];
        for (int k = 0; k < LSTM_H; ++k) s += hp[k] * w[k];
        bb[j - 40] = s;
    }
    __syncthreads();
    if (j == 0) {
        const int lbl = labels[p * 2];
        float mx = logits[0]; int am = 0;
        for (int q = 1; q < 40; ++q) if (logits[q] > mx) { mx = logits[q]; am = q; }
        float se = 0.f;
        for (int q = 0; q < 40; ++q) se += __expf(logits[q] - mx);
        const float nll = (mx + __logf(se)) - logits[lbl];
        float sl = 0.f;
        for (int q = 0; q < 2; ++q) {
            const float t = props[p * 2 + q] * (1.0f / (float)SEQ);
            const float d = fabsf(bb[q] - t);
            sl += (d < 1.f) ? 0.5f * d * d : (d - 0.5f);
        }
        out3[p * 3 + 0] = nll;
        out3[p * 3 + 1] = (am == lbl) ? 1.f : 0.f;
        out3[p * 3 + 2] = sl;
    }
}

__global__ __launch_bounds__(256) void head_stage2(
    const float* __restrict__ out3, float* __restrict__ out)
{
    __shared__ float s0[256], s1[256], s2[256];
    const int t = threadIdx.x;
    float a = 0.f, b = 0.f, c = 0.f;
    for (int p = t; p < NPROP; p += 256) {
        a += out3[p * 3 + 0]; b += out3[p * 3 + 1]; c += out3[p * 3 + 2];
    }
    s0[t] = a; s1[t] = b; s2[t] = c;
    __syncthreads();
    for (int off = 128; off > 0; off >>= 1) {
        if (t < off) { s0[t] += s0[t + off]; s1[t] += s1[t + off]; s2[t] += s2[t + off]; }
        __syncthreads();
    }
    if (t == 0) {
        out[0] = (s0[0] / (float)NPROP) / (1.f + s1[0]);
        out[1] = (s2[0] / (float)(NPROP * 2)) * 10.f;
    }
}

__global__ void zero_kernel(uint32_t* __restrict__ p, size_t n) {
    const size_t i = (size_t)blockIdx.x * blockDim.x + threadIdx.x;
    if (i < n) p[i] = 0u;
}

// ---------------------------------------------------------------------------
extern "C" void kernel_launch(void* const* d_in, const int* in_sizes, int n_in,
                              void* d_out, int out_size, void* d_ws, size_t ws_size,
                              hipStream_t stream) {
    const float* data   = (const float*)d_in[0];
    const int*   labels = (const int*)  d_in[1];
    const float* props  = (const float*)d_in[2];
    const float* Wih0   = (const float*)d_in[3];
    const float* Whh0   = (const float*)d_in[4];
    const float* b0     = (const float*)d_in[5];
    const float* Wih1   = (const float*)d_in[6];
    const float* Whh1   = (const float*)d_in[7];
    const float* b1     = (const float*)d_in[8];
    const float* Wcls   = (const float*)d_in[9];
    const float* bcls   = (const float*)d_in[10];
    const float* Wbb    = (const float*)d_in[11];
    const float* bbb    = (const float*)d_in[12];

    char* ws = (char*)d_ws;
    const size_t nIH0 = (size_t)4 * LSTM_H * CIN;      // 262144
    const size_t nHH  = (size_t)4 * LSTM_H * LSTM_H;   // 1048576
    const size_t SZf  = (size_t)NPROP * LSTM_H * 4;    // 2 MB
    const size_t SZb  = (size_t)NPROP * LSTM_H * 2;    // 1 MB

    size_t off = 0;
    bf16_t* WihB0 = (bf16_t*)(ws + off); off += nIH0 * 2;
    bf16_t* WhhB0 = (bf16_t*)(ws + off); off += nHH * 2;
    bf16_t* WihB1 = (bf16_t*)(ws + off); off += nHH * 2;
    bf16_t* WhhB1 = (bf16_t*)(ws + off); off += nHH * 2;
    char*   zbase = ws + off;
    float*  c0    = (float*)(ws + off);  off += SZf;
    float*  c1    = (float*)(ws + off);  off += SZf;
    bf16_t* h0b0  = (bf16_t*)(ws + off); off += SZb;
    bf16_t* h1b0  = (bf16_t*)(ws + off); off += SZb;
    const size_t zbytes = (size_t)(ws + off - zbase);   // c0,c1,h0b0,h1b0
    bf16_t* h0b1  = (bf16_t*)(ws + off); off += SZb;
    bf16_t* h1b1  = (bf16_t*)(ws + off); off += SZb;
    float*  h1f   = (float*)(ws + off);  off += SZf;
    float*  out3  = (float*)(ws + off);  off += (size_t)NPROP * 3 * 4;
    bf16_t* h0b[2] = { h0b0, h0b1 };
    bf16_t* h1b[2] = { h1b0, h1b1 };

    // one-time per launch: weights f32 -> bf16, zero states
    cvt_f32_to_bf16<<<(unsigned)(nIH0 / (8 * 256)), 256, 0, stream>>>(Wih0, WihB0, (int)nIH0);
    cvt_f32_to_bf16<<<(unsigned)(nHH  / (8 * 256)), 256, 0, stream>>>(Whh0, WhhB0, (int)nHH);
    cvt_f32_to_bf16<<<(unsigned)(nHH  / (8 * 256)), 256, 0, stream>>>(Wih1, WihB1, (int)nHH);
    cvt_f32_to_bf16<<<(unsigned)(nHH  / (8 * 256)), 256, 0, stream>>>(Whh1, WhhB1, (int)nHH);
    {
        const size_t nwords = zbytes / 4;
        zero_kernel<<<(unsigned)((nwords + 255) / 256), 256, 0, stream>>>(
            (uint32_t*)zbase, nwords);
    }

    const dim3 grid(16, 16), blk(256);
    for (int t = 0; t < SEQ; ++t) {
        const int rd = t & 1, wr = rd ^ 1;
        // layer 0: x = data[:, t, :] (f32), K = 128 + 512
        lstm_step_kernel<<<grid, blk, 0, stream>>>(
            data, nullptr, SEQ * CIN, t * CIN,
            h0b[rd], WihB0, WhhB0, b0, c0, h0b[wr], nullptr, CIN);
        // layer 1: x = h0(t) (bf16), K = 512 + 512
        lstm_step_kernel<<<grid, blk, 0, stream>>>(
            nullptr, h0b[wr], LSTM_H, 0,
            h1b[rd], WihB1, WhhB1, b1, c1, h1b[wr], h1f, LSTM_H);
    }
    head_stage1<<<NPROP, 64, 0, stream>>>(h1f, Wcls, bcls, Wbb, bbb, labels, props, out3);
    head_stage2<<<1, 256, 0, stream>>>(out3, (float*)d_out);
}
